// Net_15616501088726
// MI455X (gfx1250) — matrix operand order repaired and checked by reading, not compiled, not run on hardware
//
#include <hip/hip_runtime.h>

// Reference's GNN output is discarded (forward overwrites x = static_feature),
// so only the FC stack matters: [N,12] -> 64 relu -> 64 relu -> 1.
// One wave32 per 16 nodes; hidden GEMMs via v_wmma_f32_16x16x32_f16.
// All operands staged in LDS as f16 with WMMA-fragment-contiguous layouts so
// each A/B fragment is built from ds_load_b128s (no divergent scalar loads).

typedef __attribute__((ext_vector_type(16))) _Float16 v16h;
typedef __attribute__((ext_vector_type(8)))  _Float16 v8h;
typedef __attribute__((ext_vector_type(2)))  _Float16 v2h;
typedef __attribute__((ext_vector_type(8)))  float    v8f;

#define SFW 12     // static feature width
#define HID 64     // hidden width
#define NW  8      // waves per block (256 threads)
#define W1S 40     // w1 LDS row stride in halves (32 used; 80B = 20-bank lane stride)
#define W2S 72     // w2 LDS row stride in halves (64 used; 144B = 36-bank lane stride)
#define AS  40     // A-tile row stride in halves (32 used)
#define HS  72     // activation row stride in halves (64 used)

__device__ __forceinline__ v16h cat16(v8h lo, v8h hi) {
    return __builtin_shufflevector(lo, hi, 0,1,2,3,4,5,6,7,8,9,10,11,12,13,14,15);
}

__global__ __launch_bounds__(256) void
fc_stack_wmma_kernel(const float* __restrict__ sf,
                     const float* __restrict__ Wfc1, const float* __restrict__ bfc1,
                     const float* __restrict__ Wfc2, const float* __restrict__ bfc2,
                     const float* __restrict__ Wp,   const float* __restrict__ bp,
                     float* __restrict__ out, int N)
{
    __shared__ __attribute__((aligned(16))) _Float16 w1h[HID * W1S];       // Wfc1^T tiles, K padded 12->32
    __shared__ __attribute__((aligned(16))) _Float16 w2h[HID * W2S];       // Wfc2 rows (out-neuron major)
    __shared__ __attribute__((aligned(16))) _Float16 atile_s[NW][16 * AS]; // per-wave input tile (f16)
    __shared__ __attribute__((aligned(16))) _Float16 hbuf_s[NW][16 * HS];  // per-wave activations (f16)
    __shared__ float bias1[HID], bias2[HID], wp[HID];

    const int tid = threadIdx.x;

    // ---- phase 1: zero padded regions; stage Wfc2/biases (all unconditional)
    {
        unsigned* z1 = (unsigned*)w1h;
        for (int i = tid; i < HID * W1S / 2; i += 256) z1[i] = 0u;
        unsigned* za = (unsigned*)&atile_s[0][0];
        for (int i = tid; i < NW * 16 * AS / 2; i += 256) za[i] = 0u;
        for (int i = tid; i < HID * HID; i += 256) {
            const int n = i >> 6, k = i & 63;
            w2h[n * W2S + k] = (_Float16)Wfc2[i];
        }
        if (tid < HID) {
            bias1[tid] = bfc1[tid];
            bias2[tid] = bfc2[tid];
            wp[tid]    = Wp[tid];
        }
    }
    __syncthreads();
    // ---- phase 2: Wfc1 values into zeroed rows (race-free vs phase-1 zeroing)
    for (int i = tid; i < HID * SFW; i += 256) {
        const int n = i / SFW, k = i - n * SFW;
        w1h[n * W1S + k] = (_Float16)Wfc1[i];
    }
    __syncthreads();

    const int lane = tid & 31;
    const int wave = tid >> 5;
    _Float16* atile = atile_s[wave];
    _Float16* hbuf  = hbuf_s[wave];

    const int nodeBase = (blockIdx.x * NW + wave) * 16;
    if (nodeBase >= N) return;               // wave-uniform: EXEC all-1s below

    // ---- cooperative static_feature -> f16 A-tile (16 rows x 12 cols) ------
    {
        const long sfBase = (long)nodeBase * SFW;
        const long gmax   = (long)N * SFW - 2;
        for (int i = lane; i < 96; i += 32) {          // 96 float2 pairs / wave
            const int off = 2 * i;
            long g = sfBase + off;
            if (g > gmax) g = gmax;                    // clamp (no branchy load)
            const float2 v = *(const float2*)(sf + g); // 8B-aligned (off even)
            const int row = off / SFW, col = off - row * SFW; // col even
            v2h p; p[0] = (_Float16)v.x; p[1] = (_Float16)v.y;
            *(v2h*)(atile + row * AS + col) = p;
        }
    }

    // WMMA wave32 fragment coordinates (ISA 7.12.2)
    const int M  = lane & 15;                 // A row
    const int kA = (lane < 16) ? 0 : 8;       // A: halves kA..kA+7, kA+16..kA+23
    const int nB = lane & 15;                 // B/C/D column
    const int kB = (lane < 16) ? 0 : 8;       // B: halves kB..kB+7, kB+16..kB+23
    const int mB = (lane < 16) ? 0 : 8;       // C/D row base
    const int node = nodeBase + M;

    // ---- Layer 1 -----------------------------------------------------------
    v8h a1l = *(const v8h*)(atile + M * AS + kA);
    v8h a1h = *(const v8h*)(atile + M * AS + 16 + kA);
    const v16h a1 = cat16(a1l, a1h);

#pragma unroll
    for (int t = 0; t < 4; ++t) {
        const int n = t * 16 + nB;
        v8h b0 = *(const v8h*)(w1h + n * W1S + kB);
        v8h b1 = *(const v8h*)(w1h + n * W1S + kB + 16);
        const float bv = bias1[n];
        v8f c = { bv, bv, bv, bv, bv, bv, bv, bv };      // bias via C operand
        c = __builtin_amdgcn_wmma_f32_16x16x32_f16(false, a1, false, cat16(b0, b1),
                                                   (short)0, c, false, false);
#pragma unroll
        for (int r = 0; r < 8; ++r)
            hbuf[(mB + r) * HS + n] = (_Float16)fmaxf(c[r], 0.0f);
    }

    // ---- Layer 2 A fragments: contiguous b128 reads of own row -------------
    v8h x0 = *(const v8h*)(hbuf + M * HS + kA);
    v8h x1 = *(const v8h*)(hbuf + M * HS + kA + 16);
    v8h x2 = *(const v8h*)(hbuf + M * HS + kA + 32);
    v8h x3 = *(const v8h*)(hbuf + M * HS + kA + 48);
    const v16h a2lo = cat16(x0, x1);                     // K = 0..31
    const v16h a2hi = cat16(x2, x3);                     // K = 32..63

#pragma unroll
    for (int t = 0; t < 4; ++t) {
        const int n = t * 16 + nB;
        v8h b0 = *(const v8h*)(w2h + n * W2S + kB);
        v8h b1 = *(const v8h*)(w2h + n * W2S + kB + 16);
        v8h b2 = *(const v8h*)(w2h + n * W2S + 32 + kB);
        v8h b3 = *(const v8h*)(w2h + n * W2S + 32 + kB + 16);
        const float bv = bias2[n];
        v8f c = { bv, bv, bv, bv, bv, bv, bv, bv };
        c = __builtin_amdgcn_wmma_f32_16x16x32_f16(false, a2lo, false, cat16(b0, b1),
                                                   (short)0, c, false, false);
        c = __builtin_amdgcn_wmma_f32_16x16x32_f16(false, a2hi, false, cat16(b2, b3),
                                                   (short)0, c, false, false);
#pragma unroll
        for (int r = 0; r < 8; ++r)
            hbuf[(mB + r) * HS + n] = (_Float16)fmaxf(c[r], 0.0f);
    }

    // ---- Layer 3: width-1 output, fp32 dot from own LDS row ----------------
    if (lane < 16 && node < N) {
        float acc = bp[0];
#pragma unroll
        for (int kk = 0; kk < 8; ++kk) {
            const v8h h8 = *(const v8h*)(hbuf + lane * HS + kk * 8);
#pragma unroll
            for (int j = 0; j < 8; ++j)
                acc = fmaf((float)h8[j], wp[kk * 8 + j], acc);
        }
        out[node] = acc;
    }
}

extern "C" void kernel_launch(void* const* d_in, const int* in_sizes, int n_in,
                              void* d_out, int out_size, void* d_ws, size_t ws_size,
                              hipStream_t stream)
{
    (void)n_in; (void)out_size; (void)d_ws; (void)ws_size;

    // setup_inputs() order:
    // 0:x 1:edge_index 2:static_feature 3:W1l 4:b1 5:W1r 6:W2l 7:b2 8:W2r
    // 9:Wfc1 10:bfc1 11:Wfc2 12:bfc2 13:Wp 14:bp
    const float* sf   = (const float*)d_in[2];
    const float* Wfc1 = (const float*)d_in[9];
    const float* bfc1 = (const float*)d_in[10];
    const float* Wfc2 = (const float*)d_in[11];
    const float* bfc2 = (const float*)d_in[12];
    const float* Wp   = (const float*)d_in[13];
    const float* bp   = (const float*)d_in[14];
    float* out = (float*)d_out;

    const int N = in_sizes[2] / SFW;                 // 100,000
    const int nodesPerBlock = NW * 16;               // 128
    const int blocks = (N + nodesPerBlock - 1) / nodesPerBlock;

    fc_stack_wmma_kernel<<<blocks, 256, 0, stream>>>(
        sf, Wfc1, bfc1, Wfc2, bfc2, Wp, bp, out, N);
}
